// LocalAttention_51402168599284
// MI455X (gfx1250) — matrix operand order repaired and checked by reading, not compiled, not run on hardware
//
#include <hip/hip_runtime.h>

// ---------------------------------------------------------------------------
// Types matching gfx1250 WMMA builtin signatures
// ---------------------------------------------------------------------------
typedef __attribute__((ext_vector_type(16))) __bf16 v16bf;
typedef __attribute__((ext_vector_type(8)))  __bf16 v8bf;
typedef __attribute__((ext_vector_type(8)))  float  v8f;

#define SEQ     4096
#define DMODEL  1024
#define NHEADS  16
#define HEADD   64
#define WINDOW  256

__device__ __forceinline__ v8bf ldg8(const __bf16* p) { return *(const v8bf*)p; }

__device__ __forceinline__ v16bf cat16(v8bf lo, v8bf hi) {
  return __builtin_shufflevector(lo, hi, 0,1,2,3,4,5,6,7,8,9,10,11,12,13,14,15);
}

__device__ __forceinline__ v8f wmma_bf16(v16bf a, v16bf b, v8f c) {
  // D = A(16x32 bf16) * B(32x16 bf16) + C(16x16 f32)
  return __builtin_amdgcn_wmma_f32_16x16x32_bf16(
      /*neg_a=*/false, a, /*neg_b=*/false, b,
      /*c_mod=*/(short)0, c, /*reuse_a=*/false, /*reuse_b=*/false);
}

__device__ __forceinline__ float red_max16(float x) {
  x = fmaxf(x, __shfl_xor(x, 1, 32));
  x = fmaxf(x, __shfl_xor(x, 2, 32));
  x = fmaxf(x, __shfl_xor(x, 4, 32));
  x = fmaxf(x, __shfl_xor(x, 8, 32));
  return x;
}
__device__ __forceinline__ float red_sum16(float x) {
  x += __shfl_xor(x, 1, 32);
  x += __shfl_xor(x, 2, 32);
  x += __shfl_xor(x, 4, 32);
  x += __shfl_xor(x, 8, 32);
  return x;
}

// ---------------------------------------------------------------------------
// 1) fp32 -> bf16 convert (x)
// ---------------------------------------------------------------------------
__global__ void cvt_bf16_kernel(const float* __restrict__ in,
                                __bf16* __restrict__ out, int n) {
  int i = blockIdx.x * blockDim.x + threadIdx.x;
  if (i < n) out[i] = (__bf16)in[i];
}

// ---------------------------------------------------------------------------
// 2) weight transpose + convert: Wt[n][k] = bf16(W[k][n]), 1024x1024
// ---------------------------------------------------------------------------
__global__ void cvt_wT_kernel(const float* __restrict__ W,
                              __bf16* __restrict__ Wt) {
  int i = blockIdx.x * blockDim.x + threadIdx.x;   // 0 .. 1M-1, coalesced store
  int n = i >> 10;
  int k = i & 1023;
  Wt[i] = (__bf16)W[(size_t)k * DMODEL + n];
}

// ---------------------------------------------------------------------------
// 3) WMMA GEMM:  C[4096x1024] = (A_bf16[4096x1024] @ Wt_bf16^T + bias) * scale
//    Wt stored [n][k] so B-operand fragment loads are contiguous.
//    One wave computes a 16x64 tile: 4 accumulators reuse one A fragment
//    per k-step (10 b128 loads : 4 wmma). Block = 4 waves -> 16x256 of C.
//    `scale` folds the attention 1/sqrt(hd) into the Q projection epilogue
//    so the attention inner loop has no per-score multiply.
//    MODE 0: store bf16 into [H][S][64]   (Q, K)
//    MODE 1: store f32  into [S][1024]    (final output)
//    MODE 2: store bf16 into [H][64][S]   (V transposed for PV matmul)
// ---------------------------------------------------------------------------
template <int MODE>
__global__ __launch_bounds__(128, 1)
void gemm_wmma_kernel(const __bf16* __restrict__ A,
                      const __bf16* __restrict__ Bt,
                      const float* __restrict__ bias,
                      void* __restrict__ outp, float scale) {
  const int lane   = threadIdx.x & 31;
  const int wave   = threadIdx.x >> 5;
  const int lane16 = lane & 15;
  const int half   = lane >> 4;

  const int n0 = (blockIdx.x * 4 + wave) * 64;   // 64 output cols per wave
  const int m0 = blockIdx.y * 16;

  const __bf16* arow = A + (size_t)(m0 + lane16) * DMODEL;  // A: m = lane%16
  const __bf16* brow[4];
#pragma unroll
  for (int f = 0; f < 4; ++f)                               // B: n = lane%16
    brow[f] = Bt + (size_t)(n0 + 16 * f + lane16) * DMODEL + half * 16;

  const v8f zero8 = {0.f,0.f,0.f,0.f,0.f,0.f,0.f,0.f};
  v8f acc[4] = {zero8, zero8, zero8, zero8};

#pragma unroll 2
  for (int kk = 0; kk < DMODEL; kk += 32) {
    // A fragment: lanes' K window = 16*half .. 16*half+15
    v16bf a = cat16(ldg8(arow + kk + half * 16),
                    ldg8(arow + kk + half * 16 + 8));
#pragma unroll
    for (int f = 0; f < 4; ++f) {
      // B fragment: lanes' K window = 16*half .. 16*half+15
      v16bf b = cat16(ldg8(brow[f] + kk), ldg8(brow[f] + kk + 8));
      acc[f] = wmma_bf16(a, b, acc[f]);
    }
    if (kk + 32 < DMODEL) {
      __builtin_prefetch(arow + kk + 64, 0, 1);
      __builtin_prefetch(brow[0] + kk + 32, 0, 1);
      __builtin_prefetch(brow[1] + kk + 32, 0, 1);
      __builtin_prefetch(brow[2] + kk + 32, 0, 1);
      __builtin_prefetch(brow[3] + kk + 32, 0, 1);
    }
  }

#pragma unroll
  for (int f = 0; f < 4; ++f) {
    const int n  = n0 + 16 * f + lane16;
    const float bn = bias[n];
#pragma unroll
    for (int v = 0; v < 8; ++v) {
      const int row = m0 + v + 8 * half;      // D layout: m = v + 8*half
      const float val = (acc[f][v] + bn) * scale;
      if (MODE == 0) {
        const int h = n >> 6, dn = n & 63;
        ((__bf16*)outp)[((size_t)h * SEQ + row) * HEADD + dn] = (__bf16)val;
      } else if (MODE == 1) {
        ((float*)outp)[(size_t)row * DMODEL + n] = val;
      } else {
        const int h = n >> 6, dn = n & 63;
        ((__bf16*)outp)[((size_t)h * HEADD + dn) * SEQ + row] = (__bf16)val;
      }
    }
  }
}

// ---------------------------------------------------------------------------
// 4) Flash-style sliding-window attention.
//    One wave per (head, 16-query tile). 32 keys per iteration:
//      4x QK^T wmma -> online softmax (shfl_xor row reductions) ->
//      P transposed through LDS to A-layout -> 4x PV wmma (hd = 64).
//    Q (pre-scaled by 1/sqrt(hd)), K in [H][S][64] bf16; V in [H][64][S] bf16.
//    __launch_bounds__(128,1) keeps qa0/qa1 + 4 accumulators in VGPRs.
// ---------------------------------------------------------------------------
__global__ __launch_bounds__(128, 1)
void attn_window_kernel(const __bf16* __restrict__ Q,
                        const __bf16* __restrict__ K,
                        const __bf16* __restrict__ Vt,
                        __bf16* __restrict__ O) {
  __shared__ __bf16 lds[4][16 * 32];   // per-wave 16x32 bf16 P tile

  const int lane   = threadIdx.x & 31;
  const int wave   = threadIdx.x >> 5;
  const int lane16 = lane & 15;
  const int half   = lane >> 4;

  const int q0 = blockIdx.x * 16;
  const int h  = blockIdx.y * 4 + wave;   // same q0 for all waves in block

  const __bf16* Qh = Q  + (size_t)h * SEQ * HEADD;
  const __bf16* Kh = K  + (size_t)h * SEQ * HEADD;
  const __bf16* Vh = Vt + (size_t)h * HEADD * SEQ;

  // Q A-fragments for K-dim chunks 0..31 and 32..63 (loaded once)
  const __bf16* qrow = Qh + (size_t)(q0 + lane16) * HEADD;
  const v16bf qa0 = cat16(ldg8(qrow + half * 16),      ldg8(qrow + half * 16 + 8));
  const v16bf qa1 = cat16(ldg8(qrow + 32 + half * 16), ldg8(qrow + 32 + half * 16 + 8));

  const v8f zero8 = {0.f,0.f,0.f,0.f,0.f,0.f,0.f,0.f};
  v8f acc[4] = {zero8, zero8, zero8, zero8};   // 16 x 64 output tile
  float m_run[8], l_run[8];
#pragma unroll
  for (int v = 0; v < 8; ++v) { m_run[v] = -1e30f; l_run[v] = 0.f; }

  int jstart = q0 - (WINDOW - 1);
  if (jstart < 0) jstart = 0;
  const int kb0 = (jstart >> 5) << 5;

  for (int kb = kb0; kb <= q0 + 15; kb += 32) {
    float p[2][8];
#pragma unroll
    for (int t = 0; t < 2; ++t) {
      const __bf16* krow = Kh + (size_t)(kb + 16 * t + lane16) * HEADD;
      v16bf b0 = cat16(ldg8(krow + half * 16),      ldg8(krow + half * 16 + 8));
      v16bf b1 = cat16(ldg8(krow + 32 + half * 16), ldg8(krow + 32 + half * 16 + 8));
      v8f s = zero8;
      s = wmma_bf16(qa0, b0, s);
      s = wmma_bf16(qa1, b1, s);
#pragma unroll
      for (int v = 0; v < 8; ++v) {
        const int i = q0 + v + 8 * half;          // query row
        const int j = kb + 16 * t + lane16;       // key col
        const bool ok = (j <= i) && ((i - j) < WINDOW);
        p[t][v] = ok ? s[v] : -3.0e38f;           // scale pre-folded into Q
      }
    }

    // online softmax over this 32-key chunk
#pragma unroll
    for (int v = 0; v < 8; ++v) {
      float mx = red_max16(fmaxf(p[0][v], p[1][v]));
      const float mn    = fmaxf(m_run[v], mx);
      const float alpha = __expf(m_run[v] - mn);
      const float p0 = __expf(p[0][v] - mn);
      const float p1 = __expf(p[1][v] - mn);
      const float rs = red_sum16(p0 + p1);
      l_run[v] = l_run[v] * alpha + rs;
      m_run[v] = mn;
      acc[0][v] *= alpha; acc[1][v] *= alpha;
      acc[2][v] *= alpha; acc[3][v] *= alpha;
      const int m = v + 8 * half;                 // P tile row
      lds[wave][m * 32 + lane16]      = (__bf16)p0;
      lds[wave][m * 32 + 16 + lane16] = (__bf16)p1;
    }
    __syncthreads();   // uniform: all waves share q0 -> same trip count

    // reload P as 16x32 A-fragment
    const __bf16* prow = &lds[wave][lane16 * 32];
    v16bf pa = cat16(ldg8(prow + half * 16), ldg8(prow + half * 16 + 8));

    // PV: B fragments from transposed V, contiguous over key dim
#pragma unroll
    for (int f = 0; f < 4; ++f) {
      const __bf16* vcol = Vh + (size_t)(16 * f + lane16) * SEQ + kb + half * 16;
      v16bf vb = cat16(ldg8(vcol), ldg8(vcol + 8));
      acc[f] = wmma_bf16(pa, vb, acc[f]);
    }
    __syncthreads();
  }

  // epilogue: normalize and scatter to [S][1024] bf16
#pragma unroll
  for (int f = 0; f < 4; ++f) {
#pragma unroll
    for (int v = 0; v < 8; ++v) {
      const int s   = q0 + v + 8 * half;
      const int col = h * HEADD + 16 * f + lane16;
      O[(size_t)s * DMODEL + col] = (__bf16)(acc[f][v] / l_run[v]);
    }
  }
}

// ---------------------------------------------------------------------------
// Launch: x->bf16, W->bf16^T, QKV GEMMs, windowed attention, O GEMM (fp32 out)
// ---------------------------------------------------------------------------
extern "C" void kernel_launch(void* const* d_in, const int* in_sizes, int n_in,
                              void* d_out, int out_size, void* d_ws, size_t ws_size,
                              hipStream_t stream) {
  const float* x  = (const float*)d_in[0];
  const float* Wq = (const float*)d_in[1];
  const float* Wk = (const float*)d_in[2];
  const float* Wv = (const float*)d_in[3];
  const float* Wo = (const float*)d_in[4];
  const float* bq = (const float*)d_in[5];
  const float* bk = (const float*)d_in[6];
  const float* bv = (const float*)d_in[7];
  const float* bo = (const float*)d_in[8];

  char* ws = (char*)d_ws;
  const size_t MB = 1u << 20;
  __bf16* xb  = (__bf16*)(ws);              //  8 MB: x bf16       [S][D]
  __bf16* WqT = (__bf16*)(ws +  8 * MB);    //  2 MB each: Wt[n][k]
  __bf16* WkT = (__bf16*)(ws + 10 * MB);
  __bf16* WvT = (__bf16*)(ws + 12 * MB);
  __bf16* WoT = (__bf16*)(ws + 14 * MB);
  __bf16* Qb  = (__bf16*)(ws + 16 * MB);    //  8 MB: [H][S][64]  (pre-scaled)
  __bf16* Kb  = (__bf16*)(ws + 24 * MB);    //  8 MB: [H][S][64]
  __bf16* Vtb = (__bf16*)(ws + 32 * MB);    //  8 MB: [H][64][S]
  __bf16* Ab  = (__bf16*)(ws + 40 * MB);    //  8 MB: attn out [S][D]

  const int nX = SEQ * DMODEL;       // 4,194,304
  const int nW = DMODEL * DMODEL;    // 1,048,576

  cvt_bf16_kernel<<<nX / 256, 256, 0, stream>>>(x, xb, nX);
  cvt_wT_kernel<<<nW / 256, 256, 0, stream>>>(Wq, WqT);
  cvt_wT_kernel<<<nW / 256, 256, 0, stream>>>(Wk, WkT);
  cvt_wT_kernel<<<nW / 256, 256, 0, stream>>>(Wv, WvT);
  cvt_wT_kernel<<<nW / 256, 256, 0, stream>>>(Wo, WoT);

  dim3 ggrid(DMODEL / 256, SEQ / 16);      // (4, 256): 4 waves x 16x64 tiles
  gemm_wmma_kernel<0><<<ggrid, 128, 0, stream>>>(xb, WqT, bq, Qb, 0.125f);
  gemm_wmma_kernel<0><<<ggrid, 128, 0, stream>>>(xb, WkT, bk, Kb, 1.0f);
  gemm_wmma_kernel<2><<<ggrid, 128, 0, stream>>>(xb, WvT, bv, Vtb, 1.0f);

  dim3 agrid(SEQ / 16, NHEADS / 4);        // (256, 4), 4 waves/block
  attn_window_kernel<<<agrid, 128, 0, stream>>>(Qb, Kb, Vtb, Ab);

  gemm_wmma_kernel<1><<<ggrid, 128, 0, stream>>>(Ab, WoT, bo, (float*)d_out, 1.0f);
}
